// CustomVisionTransformer_49194555408706
// MI455X (gfx1250) — matrix operand order repaired and checked by reading, not compiled, not run on hardware
//
#include <hip/hip_runtime.h>
#include <hip/hip_bf16.h>
#include <stdint.h>

// ---------------------------------------------------------------------------
// ViT-Base forward + JVP on gfx1250 (wave32, WMMA f32_16x16x32_f16).
// All GEMMs (patch-embed, qkv, scores, attn*v, out-proj, fc1, fc2, head)
// go through v_wmma_f32_16x16x32_f16 with f32 accumulation. Tangent GEMMs
// needing the product rule run as DUAL (two WMMAs/K-step per accumulator).
// Each wave owns 4 M-tiles x 1 N-tile: one B fragment feeds 4 WMMAs.
// K loop is software-pipelined via manual 2x unroll with alternating stage
// buffers (no register copies): stage loads are issued one step ahead of the
// WMMAs that consume them, so waits only cover year-old loads.
// ---------------------------------------------------------------------------

typedef _Float16 half_t;
typedef __attribute__((ext_vector_type(16))) _Float16 v16h;
typedef __attribute__((ext_vector_type(8)))  _Float16 v8h;
typedef __attribute__((ext_vector_type(8)))  float    v8f;

#define B_   32
#define S_   197
#define SP_  224          // tokens padded to multiple of 32
#define D_   768
#define H_   12
#define DH_  64
#define TD_  2304         // 3*D
#define MD_  3072
#define L_   12
#define NC_  1000
#define NPATCH_ 196
#define MTOK_ (B_*SP_)    // 7168
#define MIMG_ (B_*NPATCH_)// 6272

// ------------------------- wave reductions (wave32) ------------------------
__device__ __forceinline__ float wred_sum(float v) {
#pragma unroll
  for (int o = 16; o > 0; o >>= 1) v += __shfl_xor(v, o, 32);
  return v;
}
__device__ __forceinline__ float wred_max(float v) {
#pragma unroll
  for (int o = 16; o > 0; o >>= 1) v = fmaxf(v, __shfl_xor(v, o, 32));
  return v;
}

// ------------------------- WMMA fragment loaders ---------------------------
// A (16x32 f16, row-major [M,K]): lane l row = M=l%16; halves are
// K = k0 + (l<16?0:16) + 0..15 -> one contiguous 32-byte load per lane.
__device__ __forceinline__ v16h load_a_frag(const half_t* __restrict__ A,
                                            int m, int lda, int k0, int lane) {
  return *reinterpret_cast<const v16h*>(A + (size_t)m * lda + k0 + ((lane >= 16) ? 16 : 0));
}
// B (32x16 f16) from row-major W[N,K]: lane l column N=l%16; halves are
// K = k0 + (l<16?0:16) + 0..15  -> one contiguous 32-byte load per lane.
__device__ __forceinline__ v16h load_b_frag(const half_t* __restrict__ Bm,
                                            int n, int ldb, int k0, int lane) {
  return *reinterpret_cast<const v16h*>(Bm + (size_t)n * ldb + k0 + ((lane >= 16) ? 16 : 0));
}

// load / compute helpers for one pipeline stage (4 M-tiles x 1 N-tile)
template<bool DUAL>
__device__ __forceinline__ void load_stage(
    const half_t* __restrict__ a1, const half_t* __restrict__ b1,
    const half_t* __restrict__ a2, const half_t* __restrict__ b2,
    int nld, const int (&mld)[4], int lda, int ldb, int k, int lane,
    v16h& bf, v16h (&af)[4], v16h& bfd, v16h (&afd)[4])
{
  bf = load_b_frag(b1, nld, ldb, k, lane);
#pragma unroll
  for (int t = 0; t < 4; ++t) af[t] = load_a_frag(a1, mld[t], lda, k, lane);
  if constexpr (DUAL) {
    bfd = load_b_frag(b2, nld, ldb, k, lane);
#pragma unroll
    for (int t = 0; t < 4; ++t) afd[t] = load_a_frag(a2, mld[t], lda, k, lane);
  }
}

template<bool DUAL>
__device__ __forceinline__ void compute_stage(
    v8f (&acc)[4], const v16h& bf, const v16h (&af)[4],
    const v16h& bfd, const v16h (&afd)[4])
{
#pragma unroll
  for (int t = 0; t < 4; ++t)
    acc[t] = __builtin_amdgcn_wmma_f32_16x16x32_f16(false, af[t], false, bf,
                                                    (short)0, acc[t], false, false);
  if constexpr (DUAL) {
#pragma unroll
    for (int t = 0; t < 4; ++t)
      acc[t] = __builtin_amdgcn_wmma_f32_16x16x32_f16(false, afd[t], false, bfd,
                                                      (short)0, acc[t], false, false);
  }
}

// ------------------------- batched WMMA GEMM -------------------------------
// C[M,N] = alpha*(A1*B1^T (+ A2*B2^T if DUAL)) + bias[n] (+ C if ACCUM)
// A: f16 [M,K] lda; B: f16 [N,K] ldb (weight layout). Batch z = zb*subBatch+zh
// with independent (b,h) strides so attention heads batch cleanly.
template<bool DUAL, bool ACCUM, typename OutT>
__global__ __launch_bounds__(256) void wmma_gemm_kernel(
    const half_t* __restrict__ A1, const half_t* __restrict__ B1,
    const half_t* __restrict__ A2, const half_t* __restrict__ B2,
    OutT* __restrict__ C, const float* __restrict__ bias, float alpha,
    int M, int N, int K, int lda, int ldb, int ldc, int subBatch,
    long long sAb, long long sAh, long long sBb, long long sBh,
    long long sCb, long long sCh)
{
  const int lane   = threadIdx.x & 31;
  const int wave   = threadIdx.x >> 5;
  const int mtile0 = (blockIdx.y * 8 + wave) * 4;      // 4 M-tiles per wave
  if (mtile0 * 16 >= M) return;                        // wave-uniform
  const int ntile = blockIdx.x;
  const int z  = blockIdx.z;
  const int zb = z / subBatch;
  const int zh = z - zb * subBatch;

  const half_t* a1 = A1 + (long long)zb * sAb + (long long)zh * sAh;
  const half_t* b1 = B1 + (long long)zb * sBb + (long long)zh * sBh;
  const half_t* a2 = nullptr;
  const half_t* b2 = nullptr;
  if constexpr (DUAL) {
    a2 = A2 + (long long)zb * sAb + (long long)zh * sAh;
    b2 = B2 + (long long)zb * sBb + (long long)zh * sBh;
  }
  OutT* c = C + (long long)zb * sCb + (long long)zh * sCh;

  v8f acc[4];
#pragma unroll
  for (int t = 0; t < 4; ++t)
#pragma unroll
    for (int i = 0; i < 8; ++i) acc[t][i] = 0.f;

  // clamp load indices so every fragment load is unconditional & in-bounds
  const int ncol = ntile * 16 + (lane & 15);
  const int nld  = (ncol < N) ? ncol : (N - 1);
  int mld[4];
#pragma unroll
  for (int t = 0; t < 4; ++t) {
    int m  = (mtile0 + t) * 16 + (lane & 15);
    mld[t] = (m < M) ? m : (M - 1);
  }

  // -------- software-pipelined K loop: alternating stages, no reg copies
  v16h bfA, afA[4], bfdA, afdA[4];
  v16h bfB, afB[4], bfdB, afdB[4];
  const int nk = K >> 5;                               // K multiple of 32
  load_stage<DUAL>(a1, b1, a2, b2, nld, mld, lda, ldb, 0, lane,
                   bfA, afA, bfdA, afdA);
  int i = 0;
  for (; i + 2 <= nk; i += 2) {
    load_stage<DUAL>(a1, b1, a2, b2, nld, mld, lda, ldb, (i + 1) * 32, lane,
                     bfB, afB, bfdB, afdB);
    compute_stage<DUAL>(acc, bfA, afA, bfdA, afdA);
    const int k2 = (i + 2 < nk) ? (i + 2) * 32 : (i + 1) * 32;  // clamp: L0 hit
    load_stage<DUAL>(a1, b1, a2, b2, nld, mld, lda, ldb, k2, lane,
                     bfA, afA, bfdA, afdA);
    compute_stage<DUAL>(acc, bfB, afB, bfdB, afdB);
  }
  if (i < nk)                                          // odd nk (K=224)
    compute_stage<DUAL>(acc, bfA, afA, bfdA, afdA);

  if (ncol < N) {
    const float bv   = bias ? bias[ncol] : 0.f;
    const int   moff = (lane >= 16) ? 8 : 0;
#pragma unroll
    for (int t = 0; t < 4; ++t) {
      if ((mtile0 + t) * 16 >= M) break;               // M multiple of 16
#pragma unroll
      for (int r = 0; r < 8; ++r) {
        const int mm = (mtile0 + t) * 16 + r + moff;
        OutT* cp = c + (size_t)mm * ldc + ncol;
        float v = acc[t][r] * alpha + bv;
        if constexpr (ACCUM) v += (float)(*cp);
        *cp = (OutT)v;
      }
    }
  }
}

template<bool DUAL, bool ACCUM, typename OutT>
static inline void launch_gemm(hipStream_t st,
    const half_t* A1, const half_t* B1, const half_t* A2, const half_t* B2,
    OutT* C, const float* bias, float alpha,
    int M, int N, int K, int lda, int ldb, int ldc,
    int batches, int subBatch,
    long long sAb, long long sAh, long long sBb, long long sBh,
    long long sCb, long long sCh)
{
  dim3 grid((N + 15) / 16, ((M + 15) / 16 + 31) / 32, batches);
  wmma_gemm_kernel<DUAL, ACCUM, OutT><<<grid, dim3(256), 0, st>>>(
      A1, B1, A2, B2, C, bias, alpha, M, N, K, lda, ldb, ldc, subBatch,
      sAb, sAh, sBb, sBh, sCb, sCh);
}

// ------------------------- elementwise / reduction kernels -----------------
__global__ __launch_bounds__(256) void convert_f32_f16(
    const float* __restrict__ src, half_t* __restrict__ dst, long long n)
{
  long long i = (long long)blockIdx.x * blockDim.x + threadIdx.x;
  if (i < n) dst[i] = (half_t)src[i];
}

// im2col for Conv2d(3,D,16,16): row = b*196 + g*14 + h, col = c*256+p*16+q
__global__ __launch_bounds__(256) void im2col_kernel(
    const float* __restrict__ x, const float* __restrict__ tg,
    half_t* __restrict__ XiP, half_t* __restrict__ XiT, long long n)
{
  long long i = (long long)blockIdx.x * blockDim.x + threadIdx.x;
  if (i >= n) return;
  int col = (int)(i % D_);
  long long r = i / D_;
  int t = (int)(r % NPATCH_);
  int b = (int)(r / NPATCH_);
  int g = t / 14, hh = t % 14;
  int c = col >> 8, rem = col & 255, p = rem >> 4, q = rem & 15;
  long long src = (((long long)(b * 3 + c) * 224) + g * 16 + p) * 224 + hh * 16 + q;
  XiP[i] = (half_t)x[src];
  XiT[i] = (half_t)tg[src];
}

// z0 = [cls+pos ; tok+pos ; zero pad], tangent has no cls/pos contribution
__global__ __launch_bounds__(256) void assemble_z0(
    const float* __restrict__ tokP, const float* __restrict__ tokT,
    const float* __restrict__ cls, const float* __restrict__ pos,
    float* __restrict__ zP, float* __restrict__ zT, long long n)
{
  long long i = (long long)blockIdx.x * blockDim.x + threadIdx.x;
  if (i >= n) return;
  int d = (int)(i % D_);
  long long r = i / D_;
  int s = (int)(r % SP_);
  int b = (int)(r / SP_);
  float vp = 0.f, vt = 0.f;
  if (s == 0) {
    vp = cls[d] + pos[d];
  } else if (s <= NPATCH_) {
    long long tr = ((long long)b * NPATCH_ + (s - 1)) * D_ + d;
    vp = tokP[tr] + pos[(long long)s * D_ + d];
    vt = tokT[tr];
  }
  zP[i] = vp;
  zT[i] = vt;
}

// LayerNorm + JVP; one wave per row, D=768 fixed. Pads (s>=197) write zeros.
__global__ __launch_bounds__(256) void ln_jvp_kernel(
    const float* __restrict__ zP, const float* __restrict__ zT,
    const float* __restrict__ gam, const float* __restrict__ bet,
    half_t* __restrict__ yP, half_t* __restrict__ yT,
    int numRows, long long inStride, int outLd, int spMod)
{
  int row  = blockIdx.x * 8 + (threadIdx.x >> 5);
  if (row >= numRows) return;
  int lane = threadIdx.x & 31;
  half_t* op = yP + (size_t)row * outLd;
  half_t* ot = yT + (size_t)row * outLd;
  if (spMod && (row % spMod) >= S_) {
    for (int i = lane; i < D_; i += 32) { op[i] = (half_t)0.f; ot[i] = (half_t)0.f; }
    return;
  }
  const float* xp = zP + (size_t)row * inStride;
  const float* xt = zT + (size_t)row * inStride;
  float x[24], dx[24];
  float s1 = 0.f, s2 = 0.f, sd = 0.f, sxd = 0.f;
#pragma unroll
  for (int i = 0; i < 24; ++i) {
    int idx = lane + 32 * i;
    x[i] = xp[idx]; dx[i] = xt[idx];
    s1 += x[i]; s2 += x[i] * x[i]; sd += dx[i]; sxd += x[i] * dx[i];
  }
  s1 = wred_sum(s1); s2 = wred_sum(s2); sd = wred_sum(sd); sxd = wred_sum(sxd);
  const float invD = 1.f / (float)D_;
  float mu  = s1 * invD, dmu = sd * invD;
  float var = s2 * invD - mu * mu;
  float r   = rsqrtf(var + 1e-6f);
  float cc  = sxd * invD - mu * dmu;       // mean(xc*dxc)
  float r3c = r * r * r * cc;
#pragma unroll
  for (int i = 0; i < 24; ++i) {
    int idx = lane + 32 * i;
    float g  = gam[idx];
    float xc = x[i] - mu;
    op[idx] = (half_t)(xc * r * g + bet[idx]);
    ot[idx] = (half_t)(((dx[i] - dmu) * r - xc * r3c) * g);
  }
}

// softmax + JVP over 197 keys; one wave per (b,h,q); writes f16 p, dp padded.
__global__ __launch_bounds__(256) void softmax_jvp_kernel(
    const float* __restrict__ sc, const float* __restrict__ dsc,
    half_t* __restrict__ p, half_t* __restrict__ dp)
{
  int row  = blockIdx.x * 8 + (threadIdx.x >> 5);
  if (row >= B_ * H_ * SP_) return;
  int lane = threadIdx.x & 31;
  int q    = row % SP_;
  half_t* pr  = p  + (size_t)row * SP_;
  half_t* dpr = dp + (size_t)row * SP_;
  if (q >= S_) {
    for (int i = lane; i < SP_; i += 32) { pr[i] = (half_t)0.f; dpr[i] = (half_t)0.f; }
    return;
  }
  const float* sr  = sc  + (size_t)row * SP_;
  const float* dsr = dsc + (size_t)row * SP_;
  float sv[7], dv[7];
  float mx = -1e30f;
#pragma unroll
  for (int i = 0; i < 7; ++i) {
    int idx = lane + 32 * i;
    if (idx < S_) { sv[i] = sr[idx]; dv[i] = dsr[idx]; }
    else          { sv[i] = -1e30f;  dv[i] = 0.f; }
    mx = fmaxf(mx, sv[i]);
  }
  mx = wred_max(mx);
  float sum = 0.f;
#pragma unroll
  for (int i = 0; i < 7; ++i) {
    int idx = lane + 32 * i;
    sv[i] = (idx < S_) ? __expf(sv[i] - mx) : 0.f;
    sum += sv[i];
  }
  sum = wred_sum(sum);
  float inv = 1.f / sum;
  float rdot = 0.f;
#pragma unroll
  for (int i = 0; i < 7; ++i) { sv[i] *= inv; rdot += sv[i] * dv[i]; }
  rdot = wred_sum(rdot);
#pragma unroll
  for (int i = 0; i < 7; ++i) {
    int idx = lane + 32 * i;
    pr[idx]  = (half_t)sv[i];
    dpr[idx] = (half_t)(sv[i] * (dv[i] - rdot));
  }
}

// v / dv transpose: qkv[(b*SP+s)*3D + 2D + h*64+dh] -> vT[b,h,dh,s]
__global__ __launch_bounds__(256) void vtrans_kernel(
    const half_t* __restrict__ qkvP, const half_t* __restrict__ qkvT,
    half_t* __restrict__ vTP, half_t* __restrict__ vTT, long long n)
{
  long long i = (long long)blockIdx.x * blockDim.x + threadIdx.x;
  if (i >= n) return;
  int s = (int)(i % SP_);
  long long t = i / SP_;
  int dh = (int)(t % DH_); t /= DH_;
  int h  = (int)(t % H_);
  int b  = (int)(t / H_);
  long long src = ((long long)b * SP_ + s) * TD_ + 2 * D_ + h * DH_ + dh;
  vTP[i] = qkvP[src];
  vTT[i] = qkvT[src];
}

// exact GELU + JVP: g(x)=x*Phi(x); g'(x)=Phi(x)+x*phi(x)
__global__ __launch_bounds__(256) void gelu_jvp_kernel(
    const float* __restrict__ hp, const float* __restrict__ ht,
    half_t* __restrict__ op, half_t* __restrict__ ot, long long n)
{
  long long i = (long long)blockIdx.x * blockDim.x + threadIdx.x;
  if (i >= n) return;
  float x = hp[i], dx = ht[i];
  float cdf = 0.5f * (1.f + erff(x * 0.70710678118654752f));
  float pdf = 0.39894228040143268f * __expf(-0.5f * x * x);
  op[i] = (half_t)(x * cdf);
  ot[i] = (half_t)((cdf + x * pdf) * dx);
}

// ---------------------------------------------------------------------------
extern "C" void kernel_launch(void* const* d_in, const int* in_sizes, int n_in,
                              void* d_out, int out_size, void* d_ws, size_t ws_size,
                              hipStream_t stream)
{
  (void)in_sizes; (void)n_in; (void)out_size; (void)ws_size;
  const float* x       = (const float*)d_in[0];
  const float* tg      = (const float*)d_in[1];
  const float* conv_w  = (const float*)d_in[2];
  const float* conv_b  = (const float*)d_in[3];
  const float* cls_tok = (const float*)d_in[4];
  const float* pos     = (const float*)d_in[5];
  const float* ln1_g   = (const float*)d_in[6];
  const float* ln1_b   = (const float*)d_in[7];
  const float* qkv_w   = (const float*)d_in[8];
  const float* qkv_b   = (const float*)d_in[9];
  const float* out_w   = (const float*)d_in[10];
  const float* out_b   = (const float*)d_in[11];
  const float* ln2_g   = (const float*)d_in[12];
  const float* ln2_b   = (const float*)d_in[13];
  const float* fc1_w   = (const float*)d_in[14];
  const float* fc1_b   = (const float*)d_in[15];
  const float* fc2_w   = (const float*)d_in[16];
  const float* fc2_b   = (const float*)d_in[17];
  const float* lnf_g   = (const float*)d_in[18];
  const float* lnf_b   = (const float*)d_in[19];
  const float* head_w  = (const float*)d_in[20];
  const float* head_b  = (const float*)d_in[21];
  float* outP = (float*)d_out;
  float* outT = outP + (size_t)B_ * NC_;

  char* ws = (char*)d_ws;
  size_t off = 0;
  auto alloc = [&](size_t bytes) -> char* {
    off = (off + 255) & ~(size_t)255;
    char* p = ws + off;
    off += bytes;
    return p;
  };

  // f16 weight copies (weights are [N_out, K_in] row-major == WMMA B layout)
  half_t* wQKV  = (half_t*)alloc((size_t)L_ * TD_ * D_ * 2);
  half_t* wOUT  = (half_t*)alloc((size_t)L_ * D_  * D_ * 2);
  half_t* wFC1  = (half_t*)alloc((size_t)L_ * MD_ * D_ * 2);
  half_t* wFC2  = (half_t*)alloc((size_t)L_ * D_  * MD_ * 2);
  half_t* wCONV = (half_t*)alloc((size_t)D_ * D_ * 2);
  half_t* wHEAD = (half_t*)alloc((size_t)NC_ * D_ * 2);

  // persistent activations
  float*  zP   = (float*)alloc((size_t)MTOK_ * D_ * 4);
  float*  zT   = (float*)alloc((size_t)MTOK_ * D_ * 4);
  half_t* yP   = (half_t*)alloc((size_t)MTOK_ * D_ * 2);
  half_t* yT   = (half_t*)alloc((size_t)MTOK_ * D_ * 2);
  half_t* qkvP = (half_t*)alloc((size_t)MTOK_ * TD_ * 2);
  half_t* qkvT = (half_t*)alloc((size_t)MTOK_ * TD_ * 2);
  half_t* oP   = (half_t*)alloc((size_t)MTOK_ * D_ * 2);
  half_t* oT   = (half_t*)alloc((size_t)MTOK_ * D_ * 2);
  half_t* clsP = (half_t*)alloc((size_t)B_ * D_ * 2);
  half_t* clsT = (half_t*)alloc((size_t)B_ * D_ * 2);

  // reused region: max(patch-embed, attention, MLP) = MLP = MTOK*MD*12 bytes
  char* r1 = alloc((size_t)MTOK_ * MD_ * 12);

  // --- weight conversion ---------------------------------------------------
  auto cvt = [&](const float* s, half_t* d, long long n) {
    convert_f32_f16<<<dim3((unsigned)((n + 255) / 256)), 256, 0, stream>>>(s, d, n);
  };
  cvt(qkv_w,  wQKV,  (long long)L_ * TD_ * D_);
  cvt(out_w,  wOUT,  (long long)L_ * D_  * D_);
  cvt(fc1_w,  wFC1,  (long long)L_ * MD_ * D_);
  cvt(fc2_w,  wFC2,  (long long)L_ * D_  * MD_);
  cvt(conv_w, wCONV, (long long)D_ * D_);
  cvt(head_w, wHEAD, (long long)NC_ * D_);

  // --- patch embedding (im2col + WMMA GEMM) --------------------------------
  {
    half_t* XiP  = (half_t*)r1;
    half_t* XiT  = XiP + (size_t)MIMG_ * D_;
    float*  tokP = (float*)(XiT + (size_t)MIMG_ * D_);
    float*  tokT = tokP + (size_t)MIMG_ * D_;
    long long n  = (long long)MIMG_ * D_;
    im2col_kernel<<<dim3((unsigned)((n + 255) / 256)), 256, 0, stream>>>(x, tg, XiP, XiT, n);
    launch_gemm<false, false, float>(stream, XiP, wCONV, nullptr, nullptr, tokP, conv_b, 1.f,
        MIMG_, D_, D_, D_, D_, D_, 1, 1, 0, 0, 0, 0, 0, 0);
    launch_gemm<false, false, float>(stream, XiT, wCONV, nullptr, nullptr, tokT, nullptr, 1.f,
        MIMG_, D_, D_, D_, D_, D_, 1, 1, 0, 0, 0, 0, 0, 0);
    long long nz = (long long)MTOK_ * D_;
    assemble_z0<<<dim3((unsigned)((nz + 255) / 256)), 256, 0, stream>>>(tokP, tokT, cls_tok, pos, zP, zT, nz);
  }

  // region aliases for attention / MLP phases (used sequentially per layer)
  float*  scP = (float*)r1;
  float*  scT = scP + (size_t)B_ * H_ * SP_ * SP_;
  half_t* prP = (half_t*)(scT + (size_t)B_ * H_ * SP_ * SP_);
  half_t* prT = prP + (size_t)B_ * H_ * SP_ * SP_;
  half_t* vTP = prT + (size_t)B_ * H_ * SP_ * SP_;
  half_t* vTT = vTP + (size_t)B_ * H_ * DH_ * SP_;
  float*  hpP = (float*)r1;
  float*  hpT = hpP + (size_t)MTOK_ * MD_;
  half_t* hgP = (half_t*)(hpT + (size_t)MTOK_ * MD_);
  half_t* hgT = hgP + (size_t)MTOK_ * MD_;

  const long long sQb = (long long)SP_ * TD_;          // qkv batch stride (b)
  const long long sSb = (long long)H_ * SP_ * SP_;     // scores/probs (b)
  const long long sSh = (long long)SP_ * SP_;          // scores/probs (h)
  const long long sVb = (long long)H_ * DH_ * SP_;     // vT (b)
  const long long sVh = (long long)DH_ * SP_;          // vT (h)

  for (int l = 0; l < L_; ++l) {
    const half_t* wq = wQKV + (size_t)l * TD_ * D_;
    const half_t* wo = wOUT + (size_t)l * D_  * D_;
    const half_t* w1 = wFC1 + (size_t)l * MD_ * D_;
    const half_t* w2 = wFC2 + (size_t)l * D_  * MD_;

    // LN1 (+JVP) -> f16 y
    ln_jvp_kernel<<<MTOK_ / 8, 256, 0, stream>>>(zP, zT, ln1_g + l * D_, ln1_b + l * D_,
        yP, yT, MTOK_, (long long)D_, D_, SP_);

    // QKV projection
    launch_gemm<false, false, half_t>(stream, yP, wq, nullptr, nullptr, qkvP, qkv_b + (size_t)l * TD_, 1.f,
        MTOK_, TD_, D_, D_, D_, TD_, 1, 1, 0, 0, 0, 0, 0, 0);
    launch_gemm<false, false, half_t>(stream, yT, wq, nullptr, nullptr, qkvT, nullptr, 1.f,
        MTOK_, TD_, D_, D_, D_, TD_, 1, 1, 0, 0, 0, 0, 0, 0);

    // scores = q k^T / 8 ; dscores = (dq k^T + q dk^T)/8   [batched over b,h]
    launch_gemm<false, false, float>(stream, qkvP, qkvP + D_, nullptr, nullptr, scP, nullptr, 0.125f,
        SP_, SP_, DH_, TD_, TD_, SP_, B_ * H_, H_,
        sQb, DH_, sQb, DH_, sSb, sSh);
    launch_gemm<true, false, float>(stream, qkvT, qkvP + D_, qkvP, qkvT + D_, scT, nullptr, 0.125f,
        SP_, SP_, DH_, TD_, TD_, SP_, B_ * H_, H_,
        sQb, DH_, sQb, DH_, sSb, sSh);

    // softmax + JVP -> f16 probs (key pads zeroed)
    softmax_jvp_kernel<<<(B_ * H_ * SP_) / 8, 256, 0, stream>>>(scP, scT, prP, prT);

    // transpose v, dv -> [B,H,DH,SP]
    {
      long long n = (long long)B_ * H_ * DH_ * SP_;
      vtrans_kernel<<<dim3((unsigned)((n + 255) / 256)), 256, 0, stream>>>(qkvP, qkvT, vTP, vTT, n);
    }

    // o = p v ; do = dp v + p dv
    launch_gemm<false, false, half_t>(stream, prP, vTP, nullptr, nullptr, oP, nullptr, 1.f,
        SP_, DH_, SP_, SP_, SP_, D_, B_ * H_, H_,
        sSb, sSh, sVb, sVh, (long long)SP_ * D_, (long long)DH_);
    launch_gemm<true, false, half_t>(stream, prT, vTP, prP, vTT, oT, nullptr, 1.f,
        SP_, DH_, SP_, SP_, SP_, D_, B_ * H_, H_,
        sSb, sSh, sVb, sVh, (long long)SP_ * D_, (long long)DH_);

    // out projection + residual (accumulate into f32 z)
    launch_gemm<false, true, float>(stream, oP, wo, nullptr, nullptr, zP, out_b + (size_t)l * D_, 1.f,
        MTOK_, D_, D_, D_, D_, D_, 1, 1, 0, 0, 0, 0, 0, 0);
    launch_gemm<false, true, float>(stream, oT, wo, nullptr, nullptr, zT, nullptr, 1.f,
        MTOK_, D_, D_, D_, D_, D_, 1, 1, 0, 0, 0, 0, 0, 0);

    // LN2 (+JVP)
    ln_jvp_kernel<<<MTOK_ / 8, 256, 0, stream>>>(zP, zT, ln2_g + l * D_, ln2_b + l * D_,
        yP, yT, MTOK_, (long long)D_, D_, SP_);

    // FC1 -> f32 pre-activation
    launch_gemm<false, false, float>(stream, yP, w1, nullptr, nullptr, hpP, fc1_b + (size_t)l * MD_, 1.f,
        MTOK_, MD_, D_, D_, D_, MD_, 1, 1, 0, 0, 0, 0, 0, 0);
    launch_gemm<false, false, float>(stream, yT, w1, nullptr, nullptr, hpT, nullptr, 1.f,
        MTOK_, MD_, D_, D_, D_, MD_, 1, 1, 0, 0, 0, 0, 0, 0);

    // exact GELU + JVP -> f16
    {
      long long n = (long long)MTOK_ * MD_;
      gelu_jvp_kernel<<<dim3((unsigned)((n + 255) / 256)), 256, 0, stream>>>(hpP, hpT, hgP, hgT, n);
    }

    // FC2 + residual
    launch_gemm<false, true, float>(stream, hgP, w2, nullptr, nullptr, zP, fc2_b + (size_t)l * D_, 1.f,
        MTOK_, D_, MD_, MD_, MD_, D_, 1, 1, 0, 0, 0, 0, 0, 0);
    launch_gemm<false, true, float>(stream, hgT, w2, nullptr, nullptr, zT, nullptr, 1.f,
        MTOK_, D_, MD_, MD_, MD_, D_, 1, 1, 0, 0, 0, 0, 0, 0);
  }

  // final LN on class token only (rows = B, stride SP*D) + head GEMM
  ln_jvp_kernel<<<(B_ + 7) / 8, 256, 0, stream>>>(zP, zT, lnf_g, lnf_b,
      clsP, clsT, B_, (long long)SP_ * D_, D_, 0);
  launch_gemm<false, false, float>(stream, clsP, wHEAD, nullptr, nullptr, outP, head_b, 1.f,
      B_, NC_, D_, D_, D_, NC_, 1, 1, 0, 0, 0, 0, 0, 0);
  launch_gemm<false, false, float>(stream, clsT, wHEAD, nullptr, nullptr, outT, nullptr, 1.f,
      B_, NC_, D_, D_, D_, NC_, 1, 1, 0, 0, 0, 0, 0, 0);
}
